// CrystalDiffusionBlock_83571473646095
// MI455X (gfx1250) — matrix operand order repaired and checked by reading, not compiled, not run on hardware
//
#include <hip/hip_runtime.h>
#include <hip/hip_bf16.h>

// ---------------- problem constants (match reference) ----------------
#define NN 50000   // nodes
#define EE 800000  // edges
#define DD 128     // hidden dim
#define BB 32      // graphs
#define LL 3       // layers

typedef __attribute__((ext_vector_type(16))) __bf16 v16bf;
typedef __attribute__((ext_vector_type(8)))  float  v8f;

// ---------------- WMMA fragment helpers (CDNA5, wave32) ----------------
// A fragment: 16x32 bf16 (MxK). ISA 7.12.2: lane%16 = M, lane/16 selects K-half;
// VGPR r holds K = {2r, 2r+1} for r<4 and K = {2r+8, 2r+9} for r>=4, +8 for hi lanes.
__device__ __forceinline__ v16bf frag_a(const __bf16* base, int ld, int kk, int lane) {
  int m = lane & 15, g = (lane >> 4) & 1;
  v16bf a;
#pragma unroll
  for (int i = 0; i < 16; ++i) {
    int r = i >> 1, h = i & 1;
    int k = kk * 32 + 2 * r + ((r >= 4) ? 8 : 0) + h + 8 * g;
    a[i] = base[m * ld + k];
  }
  return a;
}

// B fragment: 32x16 bf16 (KxN), row-major weights [K][ld].
// Dense B layout (matches A): element i<8 holds K=i+8g, i>=8 holds K=i+8+8g (g = lane/16); lane%16 = N.
__device__ __forceinline__ v16bf frag_b(const __bf16* w, int ld, int kk, int n0, int lane) {
  int n = lane & 15, g = (lane >> 4) & 1;
  v16bf b;
#pragma unroll
  for (int i = 0; i < 16; ++i) {
    int k = kk * 32 + ((i < 8) ? i : i + 8) + 8 * g;
    b[i] = w[k * ld + n0 + n];
  }
  return b;
}

__device__ __forceinline__ v8f wmma_bf16(v16bf a, v16bf b, v8f c) {
  // 8 args: (neg_a, A, neg_b, B, c_mod, C, reuse_a, reuse_b)
  return __builtin_amdgcn_wmma_f32_16x16x32_bf16(false, a, false, b, (short)0, c, false, false);
}

__device__ __forceinline__ float silu(float v) { return v / (1.0f + __expf(-v)); }

// ---------------- small utility kernels ----------------
__global__ void k_zero_f32(float* p, long n) {
  long i = (long)blockIdx.x * blockDim.x + threadIdx.x;
  if (i < n) p[i] = 0.0f;
}

__global__ void k_cvt_bf16(const float* src, __bf16* dst, long n) {
  long i = (long)blockIdx.x * blockDim.x + threadIdx.x;
  if (i < n) dst[i] = (__bf16)src[i];
}

// time MLP stage 1: hid = silu(emb @ w1 + b1), [B,D]
__global__ void k_time_mlp1(const float* emb, const float* w1, const float* b1, float* hid) {
  int idx = blockIdx.x * blockDim.x + threadIdx.x;
  if (idx >= BB * DD) return;
  int b = idx >> 7, d = idx & (DD - 1);
  float acc = b1[d];
  for (int k = 0; k < DD; ++k) acc += emb[b * DD + k] * w1[k * DD + d];
  hid[idx] = silu(acc);
}

// time MLP stage 2: tp = hid @ w2 + b2
__global__ void k_time_mlp2(const float* hid, const float* w2, const float* b2, float* tp) {
  int idx = blockIdx.x * blockDim.x + threadIdx.x;
  if (idx >= BB * DD) return;
  int b = idx >> 7, d = idx & (DD - 1);
  float acc = b2[d];
  for (int k = 0; k < DD; ++k) acc += hid[b * DD + k] * w2[k * DD + d];
  tp[idx] = acc;
}

// h = x + tp[batch]; also bf16 shadow copy for WMMA A operands
__global__ void k_init_h(const float* x, const float* tp, const int* batch,
                         float* h, __bf16* hb) {
  long idx = (long)blockIdx.x * blockDim.x + threadIdx.x;
  if (idx >= (long)NN * DD) return;
  int node = (int)(idx >> 7), d = (int)(idx & (DD - 1));
  float v = x[idx] + tp[batch[node] * DD + d];
  h[idx] = v;
  hb[idx] = (__bf16)v;
}

// deg[col[e]] += 1
__global__ void k_deg(const int* ei, float* deg) {
  int e = blockIdx.x * blockDim.x + threadIdx.x;
  if (e >= EE) return;
  atomicAdd(&deg[ei[EE + e]], 1.0f);
}

// dist[e] = || pos[row] - pos[col] ||
__global__ void k_dist(const float* pos, const int* ei, float* dist) {
  int e = blockIdx.x * blockDim.x + threadIdx.x;
  if (e >= EE) return;
  int r = ei[e], c = ei[EE + e];
  float dx = pos[r * 3 + 0] - pos[c * 3 + 0];
  float dy = pos[r * 3 + 1] - pos[c * 3 + 1];
  float dz = pos[r * 3 + 2] - pos[c * 3 + 2];
  dist[e] = sqrtf(dx * dx + dy * dy + dz * dz);
}

// d_out = [h (N*D) | pos (N*3)]
__global__ void k_finalize(const float* h, const float* pos, float* out) {
  long i = (long)blockIdx.x * blockDim.x + threadIdx.x;
  long nh = (long)NN * DD;
  if (i < nh) out[i] = h[i];
  else if (i < nh + (long)NN * 3) out[i] = pos[i - nh];
}

// ---------------- edge-message kernel (the FLOP hot spot) ----------------
// Weight-stationary: each block holds all B fragments + biases in VGPRs and
// grid-strides over 16-edge tiles. Per tile: vectorized gather of
// [h_col | h_row] into LDS, 8+4 bf16 WMMAs, SiLU epilogue, scatter-add.
__global__ void __launch_bounds__(256)
k_edge(const __bf16* __restrict__ hb, const int* __restrict__ ei,
       const float* __restrict__ dist,
       const __bf16* __restrict__ w1, const float* __restrict__ b1,
       const float* __restrict__ w1_dist,        // f32 row 256 of edge_w1
       const __bf16* __restrict__ w2, const float* __restrict__ b2,
       float* __restrict__ agg) {
  __shared__ __bf16 shA[16][264];   // row stride 528B (16B-aligned), padded
  __shared__ __bf16 shM[16][136];   // row stride 272B (16B-aligned), padded
  __shared__ int   scol[16];
  __shared__ int   srow[16];
  __shared__ float sdist[16];

  const int tid = threadIdx.x, lane = tid & 31, w = tid >> 5, g = (lane >> 4) & 1;
  const int n0 = w * 16, n = lane & 15;

  __builtin_prefetch(w1, 0, 3);   // global_prefetch_b8: warm weight path

  // ---- hoisted, loop-invariant weight fragments (stay resident in VGPRs) ----
  v16bf B1[8];
#pragma unroll
  for (int kk = 0; kk < 8; ++kk) B1[kk] = frag_b(w1, DD, kk, n0, lane);
  v16bf B2[4];
#pragma unroll
  for (int kk = 0; kk < 4; ++kk) B2[kk] = frag_b(w2, DD, kk, n0, lane);
  const float bias1 = b1[n0 + n];
  const float wd    = w1_dist[n0 + n];
  const float bias2 = b2[n0 + n];

  const int ntiles = EE / 16;
  for (int tile = blockIdx.x; tile < ntiles; tile += gridDim.x) {
    __syncthreads();                 // LDS reuse fence across iterations
    const int e0 = tile * 16;
    if (tid < 16) {
      int e = e0 + tid;
      srow[tid]  = ei[e];        // source j
      scol[tid]  = ei[EE + e];   // target i
      sdist[tid] = dist[e];
    }
    __syncthreads();

    // stage A-panel: 16 edges x 256 bf16, 16B vector copies (b128 load/store)
    for (int i = tid; i < 16 * 32; i += 256) {
      int m = i >> 5, s = i & 31;            // s: 0-15 -> h[col], 16-31 -> h[row]
      int node = (s < 16) ? scol[m] : srow[m];
      int k4 = s & 15;
      const uint4 v = ((const uint4*)(hb + (size_t)node * DD))[k4];
      ((uint4*)&shA[m][(s & 16) * 8])[k4] = v;
    }
    __syncthreads();

    // GEMM1: [16x256] x [256x16-tile], f32 accum
    v8f c = {};
#pragma unroll
    for (int kk = 0; kk < 8; ++kk)
      c = wmma_bf16(frag_a(&shA[0][0], 264, kk, lane), B1[kk], c);

    // + bias + dist * w1[256,:], SiLU, reshape C-layout -> A-layout via LDS
#pragma unroll
    for (int r = 0; r < 8; ++r) {
      int m = r + 8 * g;                     // C layout: vgpr r, lane half g
      shM[m][n0 + n] = (__bf16)silu(c[r] + bias1 + sdist[m] * wd);
    }
    __syncthreads();

    // GEMM2: [16x128] x [128x16-tile]
    v8f c2 = {};
#pragma unroll
    for (int kk = 0; kk < 4; ++kk)
      c2 = wmma_bf16(frag_a(&shM[0][0], 136, kk, lane), B2[kk], c2);

    // scatter-add msg to agg[col]
#pragma unroll
    for (int r = 0; r < 8; ++r) {
      int m = r + 8 * g;
      atomicAdd(&agg[(size_t)scol[m] * DD + n0 + n], c2[r] + bias2);
    }
  }
}

// ---------------- node MLP + mean-agg + residual + LayerNorm ----------------
// Weight-stationary, grid-strides over 16-node tiles.
__global__ void __launch_bounds__(256)
k_node(const float* __restrict__ h_in, const __bf16* __restrict__ hb_in,
       const float* __restrict__ agg, const float* __restrict__ deg,
       const __bf16* __restrict__ w1, const float* __restrict__ b1,
       const __bf16* __restrict__ w2, const float* __restrict__ b2,
       const float* __restrict__ lng, const float* __restrict__ lnb,
       float* __restrict__ h_out, __bf16* __restrict__ hb_out) {
  __shared__ __bf16 shA[16][136];
  __shared__ __bf16 shM[16][136];
  __shared__ float  shP[16][132];   // pre-layernorm values
  __shared__ float  shG[DD], shB[DD];

  const int tid = threadIdx.x, lane = tid & 31, w = tid >> 5, g = (lane >> 4) & 1;
  const int n0 = w * 16, n = lane & 15;

  if (tid < DD) { shG[tid] = lng[tid]; shB[tid] = lnb[tid]; }

  // hoisted weight fragments + biases
  v16bf B1[4], B2[4];
#pragma unroll
  for (int kk = 0; kk < 4; ++kk) B1[kk] = frag_b(w1, DD, kk, n0, lane);
#pragma unroll
  for (int kk = 0; kk < 4; ++kk) B2[kk] = frag_b(w2, DD, kk, n0, lane);
  const float bias1 = b1[n0 + n];
  const float bias2 = b2[n0 + n];

  const int ntiles = NN / 16;
  for (int tile = blockIdx.x; tile < ntiles; tile += gridDim.x) {
    __syncthreads();
    const int m0 = tile * 16;

    {   // stage 16 node rows, one 16B chunk per thread
      int m = tid >> 4, k4 = tid & 15;
      ((uint4*)&shA[m][0])[k4] = ((const uint4*)(hb_in + (size_t)(m0 + m) * DD))[k4];
    }
    __syncthreads();

    v8f c = {};
#pragma unroll
    for (int kk = 0; kk < 4; ++kk)
      c = wmma_bf16(frag_a(&shA[0][0], 136, kk, lane), B1[kk], c);

#pragma unroll
    for (int r = 0; r < 8; ++r) {
      int m = r + 8 * g;
      shM[m][n0 + n] = (__bf16)silu(c[r] + bias1);
    }
    __syncthreads();

    v8f c2 = {};
#pragma unroll
    for (int kk = 0; kk < 4; ++kk)
      c2 = wmma_bf16(frag_a(&shM[0][0], 136, kk, lane), B2[kk], c2);

#pragma unroll
    for (int r = 0; r < 8; ++r) {
      int m = r + 8 * g;
      int node = m0 + m;
      float dg = fmaxf(deg[node], 1.0f);
      float hn = c2[r] + bias2 + agg[(size_t)node * DD + n0 + n] / dg;
      shP[m][n0 + n] = h_in[(size_t)node * DD + n0 + n] + hn;   // h + h_new
    }
    __syncthreads();

    // per-row LayerNorm: 16 rows handled by first 16 threads
    if (tid < 16) {
      int m = tid;
      float mean = 0.0f;
      for (int d = 0; d < DD; ++d) mean += shP[m][d];
      mean *= (1.0f / DD);
      float var = 0.0f;
      for (int d = 0; d < DD; ++d) { float t = shP[m][d] - mean; var += t * t; }
      var *= (1.0f / DD);
      float rs = rsqrtf(var + 1e-5f);
      for (int d = 0; d < DD; ++d) {
        float o = (shP[m][d] - mean) * rs * shG[d] + shB[d];
        long idx = (long)(m0 + m) * DD + d;
        h_out[idx] = o;
        hb_out[idx] = (__bf16)o;
      }
    }
  }
}

// ---------------- host launcher ----------------
extern "C" void kernel_launch(void* const* d_in, const int* in_sizes, int n_in,
                              void* d_out, int out_size, void* d_ws, size_t ws_size,
                              hipStream_t stream) {
  const float* x        = (const float*)d_in[0];
  const float* pos      = (const float*)d_in[1];
  const float* time_emb = (const float*)d_in[2];
  const float* time_w1  = (const float*)d_in[3];
  const float* time_b1  = (const float*)d_in[4];
  const float* time_w2  = (const float*)d_in[5];
  const float* time_b2  = (const float*)d_in[6];
  const float* node_w1  = (const float*)d_in[7];
  const float* node_b1  = (const float*)d_in[8];
  const float* node_w2  = (const float*)d_in[9];
  const float* node_b2  = (const float*)d_in[10];
  const float* edge_w1  = (const float*)d_in[11];
  const float* edge_b1  = (const float*)d_in[12];
  const float* edge_w2  = (const float*)d_in[13];
  const float* edge_b2  = (const float*)d_in[14];
  const float* ln_g     = (const float*)d_in[15];
  const float* ln_b     = (const float*)d_in[16];
  const int*   ei       = (const int*)d_in[17];   // [2, E] (int32 under default JAX)
  const int*   batch    = (const int*)d_in[18];

  // workspace carve-out (256B aligned slices)
  char* p = (char*)d_ws;
  auto alloc = [&](size_t bytes) -> void* {
    void* r = (void*)p;
    p += (bytes + 255) & ~(size_t)255;
    return r;
  };
  float* h_a   = (float*)alloc((size_t)NN * DD * 4);
  float* h_b   = (float*)alloc((size_t)NN * DD * 4);
  float* agg   = (float*)alloc((size_t)NN * DD * 4);
  float* deg   = (float*)alloc((size_t)NN * 4);
  float* dist  = (float*)alloc((size_t)EE * 4);
  float* hid   = (float*)alloc((size_t)BB * DD * 4);
  float* tp    = (float*)alloc((size_t)BB * DD * 4);
  __bf16* hb   = (__bf16*)alloc((size_t)NN * DD * 2);
  __bf16* we1  = (__bf16*)alloc((size_t)LL * (2 * DD + 1) * DD * 2);
  __bf16* we2  = (__bf16*)alloc((size_t)LL * DD * DD * 2);
  __bf16* wn1  = (__bf16*)alloc((size_t)LL * DD * DD * 2);
  __bf16* wn2  = (__bf16*)alloc((size_t)LL * DD * DD * 2);

  const int T = 256;
  auto blocks = [](long n, int t) { return (int)((n + t - 1) / t); };

  // weights -> bf16 (once per launch; deterministic)
  k_cvt_bf16<<<blocks((long)LL * 257 * DD, T), T, 0, stream>>>(edge_w1, we1, (long)LL * 257 * DD);
  k_cvt_bf16<<<blocks((long)LL * DD * DD, T), T, 0, stream>>>(edge_w2, we2, (long)LL * DD * DD);
  k_cvt_bf16<<<blocks((long)LL * DD * DD, T), T, 0, stream>>>(node_w1, wn1, (long)LL * DD * DD);
  k_cvt_bf16<<<blocks((long)LL * DD * DD, T), T, 0, stream>>>(node_w2, wn2, (long)LL * DD * DD);

  // time projection, h init
  k_time_mlp1<<<blocks(BB * DD, T), T, 0, stream>>>(time_emb, time_w1, time_b1, hid);
  k_time_mlp2<<<blocks(BB * DD, T), T, 0, stream>>>(hid, time_w2, time_b2, tp);
  k_init_h<<<blocks((long)NN * DD, T), T, 0, stream>>>(x, tp, batch, h_a, hb);

  // degree + distances
  k_zero_f32<<<blocks(NN, T), T, 0, stream>>>(deg, NN);
  k_deg<<<blocks(EE, T), T, 0, stream>>>(ei, deg);
  k_dist<<<blocks(EE, T), T, 0, stream>>>(pos, ei, dist);

  float* hc = h_a;
  float* hn = h_b;
  for (int l = 0; l < LL; ++l) {
    k_zero_f32<<<blocks((long)NN * DD, T), T, 0, stream>>>(agg, (long)NN * DD);
    k_edge<<<4096, 256, 0, stream>>>(
        hb, ei, dist,
        we1 + (size_t)l * 257 * DD, edge_b1 + l * DD,
        edge_w1 + (size_t)l * 257 * DD + (size_t)256 * DD,   // dist row (f32)
        we2 + (size_t)l * DD * DD, edge_b2 + l * DD,
        agg);
    k_node<<<1024, 256, 0, stream>>>(
        hc, hb, agg, deg,
        wn1 + (size_t)l * DD * DD, node_b1 + l * DD,
        wn2 + (size_t)l * DD * DD, node_b2 + l * DD,
        ln_g + l * DD, ln_b + l * DD,
        hn, hb);
    float* t = hc; hc = hn; hn = t;
  }

  k_finalize<<<blocks((long)NN * DD + (long)NN * 3, T), T, 0, stream>>>(hc, pos, (float*)d_out);
}
